// CausalSelfAttention_4887672783606
// MI455X (gfx1250) — matrix operand order repaired and checked by reading, not compiled, not run on hardware
//
#include <hip/hip_runtime.h>
#include <hip/hip_bf16.h>

// ---- problem constants ----
constexpr int BB = 4;      // batch
constexpr int TT = 2048;   // sequence
constexpr int CC = 1024;   // channels
constexpr int HH = 16;     // heads
constexpr int DD = 64;     // head dim

typedef _Float16 half4  __attribute__((ext_vector_type(4)));
typedef _Float16 half8  __attribute__((ext_vector_type(8)));
typedef _Float16 half16 __attribute__((ext_vector_type(16)));
typedef float    float8 __attribute__((ext_vector_type(8)));
typedef unsigned int u32x4 __attribute__((ext_vector_type(4)));
typedef int  i32x4 __attribute__((ext_vector_type(4)));
typedef int  i32x8 __attribute__((ext_vector_type(8)));

// ---------------------------------------------------------------------------
// Tensor Data Mover: async DMA of a 2-D f16 tile (tile_d0 contiguous elements
// x tile_d1 rows) global -> LDS, rows packed contiguously.  ISA §8.3-8.5.
// Tracked by TENSORcnt; pair with __builtin_amdgcn_s_wait_tensorcnt.
// ---------------------------------------------------------------------------
__device__ __forceinline__ void tdm_load_tile_f16(
    unsigned int lds_off_bytes, const void* gaddr,
    unsigned int tile_d0, unsigned int tile_d1,
    unsigned int tensor_d0, unsigned int tensor_d1,
    unsigned long long stride0_elems) {
  unsigned long long ga = (unsigned long long)gaddr;
  u32x4 g0;
  g0[0] = 1u;                                   // count=1, user descriptor
  g0[1] = lds_off_bytes;                        // lds_addr
  g0[2] = (unsigned int)ga;                     // global_addr[31:0]
  g0[3] = ((unsigned int)(ga >> 32) & 0x01FFFFFFu) | 0x80000000u; // [56:32] | type=2
  i32x8 g1;
  g1[0] = (int)(1u << 16);                      // data_size=1 (2 bytes); mask=0
  g1[1] = (int)((tensor_d0 & 0xFFFFu) << 16);   // tensor_dim0 low16 @ 63:48
  g1[2] = (int)((tensor_d0 >> 16) | ((tensor_d1 & 0xFFFFu) << 16));
  g1[3] = (int)((tensor_d1 >> 16) | (tile_d0 << 16));   // tile_dim0 @ 127:112
  g1[4] = (int)(tile_d1 & 0xFFFFu);             // tile_dim1; tile_dim2=0 (2-D)
  g1[5] = (int)(unsigned int)stride0_elems;     // tensor_dim0_stride low32
  g1[6] = (int)((unsigned int)(stride0_elems >> 32) & 0xFFFFu);
  g1[7] = 0;
  i32x4 z4 = {0, 0, 0, 0};
#if __has_include(<hip/amd_detail/amd_gfx1250_TDM.h>)
  i32x8 z8 = {0, 0, 0, 0, 0, 0, 0, 0};
  __builtin_amdgcn_tensor_load_to_lds(g0, g1, z4, z4, z8, 0);
#else
  __builtin_amdgcn_tensor_load_to_lds(g0, g1, z4, z4, 0);
#endif
}

__device__ __forceinline__ unsigned int lds_off(const void* p) {
  return (unsigned int)(unsigned long long)p;   // low 32 bits = LDS byte offset
}

// ---------------------------------------------------------------------------
// WMMA fragment helpers (gfx1250 wave32 layouts, 16x16x32 f16 -> f32)
// A (16x32): lane l: M=l&15, holds K = ks*8+{0..7} and 16+ks*8+{0..7}, ks=l>>4
// B (32x16): lane l: N=l&15, holds K = ks*8+{0..7} and 16+ks*8+{0..7}, ks=l>>4
// C/D:       VGPR i, lane l: M = i + 8*(l>>4), N = l&15
// ---------------------------------------------------------------------------
__device__ __forceinline__ half16 load_fragA(const _Float16* src, int ld, int row0, int k0) {
  const int l = threadIdx.x & 31;
  const int m = l & 15, ks = l >> 4;
  const _Float16* p = src + (size_t)(row0 + m) * ld + k0 + ks * 8;
  half8 lo = *(const half8*)p;
  half8 hi = *(const half8*)(p + 16);
  half16 r;
#pragma unroll
  for (int i = 0; i < 8; ++i) { r[i] = lo[i]; r[i + 8] = hi[i]; }
  return r;
}

// Bsrc is "N-major": Bsrc[n*ld + k] == B[k][n]
__device__ __forceinline__ half16 load_fragB(const _Float16* src, int ld, int n0, int k0) {
  const int l = threadIdx.x & 31;
  const int n = l & 15, ks = l >> 4;
  const _Float16* p = src + (size_t)(n0 + n) * ld + k0 + ks * 8;
  half8 lo = *(const half8*)p;
  half8 hi = *(const half8*)(p + 16);
  half16 r;
#pragma unroll
  for (int i = 0; i < 8; ++i) { r[i] = lo[i]; r[i + 8] = hi[i]; }
  return r;
}

__device__ __forceinline__ float8 wmma_f16(half16 a, half16 b, float8 c) {
  return __builtin_amdgcn_wmma_f32_16x16x32_f16(false, a, false, b, (short)0, c, false, false);
}

__device__ __forceinline__ float rowmax16(float v) {
#pragma unroll
  for (int m = 8; m >= 1; m >>= 1) v = fmaxf(v, __shfl_xor(v, m, 32));
  return v;
}
__device__ __forceinline__ float rowsum16(float v) {
#pragma unroll
  for (int m = 8; m >= 1; m >>= 1) v += __shfl_xor(v, m, 32);
  return v;
}

// ---------------------------------------------------------------------------
// Kernel 0: fp32 -> f16 conversion
// ---------------------------------------------------------------------------
__global__ void csa_cvt_f16(const float* __restrict__ s, _Float16* __restrict__ d, int n4) {
  int i = blockIdx.x * blockDim.x + threadIdx.x;
  if (i < n4) {
    float4 v = ((const float4*)s)[i];
    half4 h = {(_Float16)v.x, (_Float16)v.y, (_Float16)v.z, (_Float16)v.w};
    ((half4*)d)[i] = h;
  }
}

// ---------------------------------------------------------------------------
// GEMM compute step shared by kernels 1/3: one 64-wide K-slab from LDS.
// Wave computes a 32x64 subtile: 16 WMMAs per slab (2 A-frags x 4 B-frags x 2 k).
// ---------------------------------------------------------------------------
__device__ __forceinline__ void gemm_slab(const _Float16* sA, const _Float16* sB,
                                          int wm, int wn, float8 acc[2][4]) {
#pragma unroll
  for (int kk = 0; kk < 64; kk += 32) {
    const half16 a0 = load_fragA(sA, 64, wm, kk);
    const half16 a1 = load_fragA(sA, 64, wm + 16, kk);
#pragma unroll
    for (int j = 0; j < 4; ++j) {
      const half16 b = load_fragB(sB, 64, wn + j * 16, kk);
      acc[0][j] = wmma_f16(a0, b, acc[0][j]);
      acc[1][j] = wmma_f16(a1, b, acc[1][j]);
    }
  }
}

// ---------------------------------------------------------------------------
// Kernel 1: fused QKV projection, TDM-staged block GEMM.
// Block = 256 thr (8 waves) -> 128x128 tile; K slabs of 64, double-buffered.
// q,k stored [B,H,T,D] f16; v stored transposed [B,H,D,T] f16.
// ---------------------------------------------------------------------------
__global__ __launch_bounds__(256) void csa_qkv(
    const _Float16* __restrict__ xh,
    const _Float16* __restrict__ wqh, const _Float16* __restrict__ wkh,
    const _Float16* __restrict__ wvh,
    const float* __restrict__ bq, const float* __restrict__ bk,
    const float* __restrict__ bv,
    _Float16* __restrict__ qh, _Float16* __restrict__ kh,
    _Float16* __restrict__ vth) {
  __shared__ _Float16 sA[2][128 * 64];   // 16 KB each
  __shared__ _Float16 sB[2][128 * 64];   // total 64 KB

  const int mat = blockIdx.x / 512;      // 3 matrices
  const int rem = blockIdx.x % 512;
  const int row0 = (rem >> 3) * 128;     // 64 row tiles
  const int col0 = (rem & 7) * 128;      // 8 col tiles

  const _Float16* W = (mat == 0) ? wqh : (mat == 1) ? wkh : wvh;
  const float* bias = (mat == 0) ? bq : (mat == 1) ? bk : bv;

  const int wave = threadIdx.x >> 5;
  const int wm = (wave & 3) * 32;
  const int wn = (wave >> 2) * 64;

  if (wave == 0) {
    tdm_load_tile_f16(lds_off(&sA[0][0]), xh + (size_t)row0 * CC, 64, 128,
                      CC, BB * TT, CC);
    tdm_load_tile_f16(lds_off(&sB[0][0]), W + (size_t)col0 * CC, 64, 128,
                      CC, CC, CC);
    __builtin_amdgcn_s_wait_tensorcnt(0);
  }
  __syncthreads();

  float8 acc[2][4] = {};
  int cur = 0;
  for (int k0 = 0; k0 < CC; k0 += 64) {  // 16 slabs
    if (wave == 0 && k0 + 64 < CC) {     // overlap next DMA with compute
      tdm_load_tile_f16(lds_off(&sA[cur ^ 1][0]), xh + (size_t)row0 * CC + k0 + 64,
                        64, 128, CC, BB * TT, CC);
      tdm_load_tile_f16(lds_off(&sB[cur ^ 1][0]), W + (size_t)col0 * CC + k0 + 64,
                        64, 128, CC, CC, CC);
    }
    gemm_slab(sA[cur], sB[cur], wm, wn, acc);
    __syncthreads();                     // all done reading cur
    if (wave == 0) __builtin_amdgcn_s_wait_tensorcnt(0);
    __syncthreads();                     // next buffer published
    cur ^= 1;
  }

  const int l = threadIdx.x & 31;
  const int n = l & 15, grp = l >> 4;
#pragma unroll
  for (int ai = 0; ai < 2; ++ai) {
#pragma unroll
    for (int j = 0; j < 4; ++j) {
      const int o = col0 + wn + j * 16 + n;
      const int h = o >> 6, d = o & 63;
      const float bv_ = bias[o];
#pragma unroll
      for (int i = 0; i < 8; ++i) {
        const int r = row0 + wm + ai * 16 + i + 8 * grp;
        const int b = r >> 11, t = r & (TT - 1);
        const float val = acc[ai][j][i] + bv_;
        if (mat == 2)
          vth[(((size_t)b * HH + h) * DD + d) * TT + t] = (_Float16)val;
        else if (mat == 1)
          kh[(((size_t)b * HH + h) * TT + t) * DD + d] = (_Float16)val;
        else
          qh[(((size_t)b * HH + h) * TT + t) * DD + d] = (_Float16)val;
      }
    }
  }
}

// ---------------------------------------------------------------------------
// Kernel 2: attention, online softmax, 64-key pipeline stages.
// Block = 4 waves sharing one (b,h); K (64x64) and Vt (64x64) tiles staged by
// TDM, double-buffered.  Per-wave 16x64 P tile bounces through LDS.
// ---------------------------------------------------------------------------
__global__ __launch_bounds__(128) void csa_attn(
    const _Float16* __restrict__ qh, const _Float16* __restrict__ kh,
    const _Float16* __restrict__ vth, _Float16* __restrict__ yh) {
  __shared__ _Float16 sK[2][64 * 64];    // [s][d]   8 KB each
  __shared__ _Float16 sV[2][64 * 64];    // [d][s]   8 KB each
  __shared__ _Float16 pbuf[4][16 * 64];  // 2 KB per wave

  const int wave = threadIdx.x >> 5;
  const int wid = blockIdx.x * 4 + wave;
  const int bh = wid >> 7;               // all 4 waves share (b,h)
  const int t0 = (wid & 127) * 16;
  const int b = bh >> 4, h = bh & 15;

  const _Float16* qb = qh + (size_t)bh * TT * DD;
  const _Float16* kb = kh + (size_t)bh * TT * DD;
  const _Float16* vb = vth + (size_t)bh * DD * TT;
  _Float16* pw = pbuf[wave];

  const int l = threadIdx.x & 31;
  const int n = l & 15, grp = l >> 4;
  const float scale = 0.125f;            // 1/sqrt(64)

  const half16 qa0 = load_fragA(qb, DD, t0, 0);
  const half16 qa1 = load_fragA(qb, DD, t0, 32);

  if (wave == 0) {
    tdm_load_tile_f16(lds_off(&sK[0][0]), kb, 64, 64, DD, TT, DD);
    tdm_load_tile_f16(lds_off(&sV[0][0]), vb, 64, 64, TT, DD, TT);
    __builtin_amdgcn_s_wait_tensorcnt(0);
  }
  __syncthreads();

  float m_i[8], l_i[8];
  float8 acc[4] = {};
#pragma unroll
  for (int i = 0; i < 8; ++i) { m_i[i] = -3.0e38f; l_i[i] = 0.0f; }

  int cur = 0;
  for (int s0 = 0; s0 < TT; s0 += 64) {  // 32 stages
    if (wave == 0 && s0 + 64 < TT) {
      tdm_load_tile_f16(lds_off(&sK[cur ^ 1][0]), kb + (size_t)(s0 + 64) * DD,
                        64, 64, DD, TT, DD);
      tdm_load_tile_f16(lds_off(&sV[cur ^ 1][0]), vb + (size_t)(s0 + 64),
                        64, 64, TT, DD, TT);
    }

    // S: 4 column tiles of 16 keys each (8 WMMAs)
    float8 S[4];
#pragma unroll
    for (int c = 0; c < 4; ++c) {
      float8 s = {};
      s = wmma_f16(qa0, load_fragB(sK[cur], 64, c * 16, 0), s);
      s = wmma_f16(qa1, load_fragB(sK[cur], 64, c * 16, 32), s);
      S[c] = s;
    }

    float mnew[8], alpha[8];
#pragma unroll
    for (int i = 0; i < 8; ++i) {
      float mx = S[0][i] * scale;
#pragma unroll
      for (int c = 0; c < 4; ++c) {
        S[c][i] *= scale;
        mx = fmaxf(mx, S[c][i]);
      }
      mnew[i] = fmaxf(m_i[i], rowmax16(mx));
    }
#pragma unroll
    for (int i = 0; i < 8; ++i) {
      alpha[i] = __expf(m_i[i] - mnew[i]);
      const int row = i + 8 * grp;
      float psum = 0.0f;
#pragma unroll
      for (int c = 0; c < 4; ++c) {
        const float p = __expf(S[c][i] - mnew[i]);
        psum += p;
        pw[row * 64 + c * 16 + n] = (_Float16)p;
      }
      l_i[i] = l_i[i] * alpha[i] + rowsum16(psum);
      m_i[i] = mnew[i];
    }
#pragma unroll
    for (int j = 0; j < 4; ++j)
#pragma unroll
      for (int i = 0; i < 8; ++i) acc[j][i] *= alpha[i];

    // wave-local LDS fence for the P round-trip (C-layout -> A-layout)
    asm volatile("s_wait_dscnt 0" ::: "memory");
    const half16 pa0 = load_fragA(pw, 64, 0, 0);
    const half16 pa1 = load_fragA(pw, 64, 0, 32);

    // PV: 8 WMMAs
#pragma unroll
    for (int j = 0; j < 4; ++j) {
      acc[j] = wmma_f16(pa0, load_fragB(sV[cur], 64, j * 16, 0), acc[j]);
      acc[j] = wmma_f16(pa1, load_fragB(sV[cur], 64, j * 16, 32), acc[j]);
    }

    __syncthreads();                     // all waves done with cur K/V
    if (wave == 0) __builtin_amdgcn_s_wait_tensorcnt(0);
    __syncthreads();                     // next K/V published
    cur ^= 1;
  }

#pragma unroll
  for (int i = 0; i < 8; ++i) {
    const float rl = 1.0f / l_i[i];
    const int t = t0 + i + 8 * grp;
#pragma unroll
    for (int j = 0; j < 4; ++j) {
      yh[((size_t)b * TT + t) * CC + h * 64 + j * 16 + n] =
          (_Float16)(acc[j][i] * rl);
    }
  }
}

// ---------------------------------------------------------------------------
// Kernel 3: output projection  out = y @ Wp^T + bp  (fp32 out), TDM-staged.
// ---------------------------------------------------------------------------
__global__ __launch_bounds__(256) void csa_proj(
    const _Float16* __restrict__ yh, const _Float16* __restrict__ wph,
    const float* __restrict__ bp, float* __restrict__ out) {
  __shared__ _Float16 sA[2][128 * 64];
  __shared__ _Float16 sB[2][128 * 64];

  const int row0 = (blockIdx.x >> 3) * 128;
  const int col0 = (blockIdx.x & 7) * 128;
  const int wave = threadIdx.x >> 5;
  const int wm = (wave & 3) * 32;
  const int wn = (wave >> 2) * 64;

  if (wave == 0) {
    tdm_load_tile_f16(lds_off(&sA[0][0]), yh + (size_t)row0 * CC, 64, 128,
                      CC, BB * TT, CC);
    tdm_load_tile_f16(lds_off(&sB[0][0]), wph + (size_t)col0 * CC, 64, 128,
                      CC, CC, CC);
    __builtin_amdgcn_s_wait_tensorcnt(0);
  }
  __syncthreads();

  float8 acc[2][4] = {};
  int cur = 0;
  for (int k0 = 0; k0 < CC; k0 += 64) {
    if (wave == 0 && k0 + 64 < CC) {
      tdm_load_tile_f16(lds_off(&sA[cur ^ 1][0]), yh + (size_t)row0 * CC + k0 + 64,
                        64, 128, CC, BB * TT, CC);
      tdm_load_tile_f16(lds_off(&sB[cur ^ 1][0]), wph + (size_t)col0 * CC + k0 + 64,
                        64, 128, CC, CC, CC);
    }
    gemm_slab(sA[cur], sB[cur], wm, wn, acc);
    __syncthreads();
    if (wave == 0) __builtin_amdgcn_s_wait_tensorcnt(0);
    __syncthreads();
    cur ^= 1;
  }

  const int l = threadIdx.x & 31;
  const int n = l & 15, grp = l >> 4;
#pragma unroll
  for (int ai = 0; ai < 2; ++ai) {
#pragma unroll
    for (int j = 0; j < 4; ++j) {
      const int o = col0 + wn + j * 16 + n;
      const float bv_ = bp[o];
#pragma unroll
      for (int i = 0; i < 8; ++i) {
        const int r = row0 + wm + ai * 16 + i + 8 * grp;
        out[(size_t)r * CC + o] = acc[ai][j][i] + bv_;
      }
    }
  }
}

// ---------------------------------------------------------------------------
extern "C" void kernel_launch(void* const* d_in, const int* in_sizes, int n_in,
                              void* d_out, int out_size, void* d_ws, size_t ws_size,
                              hipStream_t stream) {
  const float* x  = (const float*)d_in[0];
  const float* Wq = (const float*)d_in[1];
  const float* bq = (const float*)d_in[2];
  const float* Wk = (const float*)d_in[3];
  const float* bk = (const float*)d_in[4];
  const float* Wv = (const float*)d_in[5];
  const float* bv = (const float*)d_in[6];
  const float* Wp = (const float*)d_in[7];
  const float* bp = (const float*)d_in[8];
  float* out = (float*)d_out;

  const size_t NX = (size_t)BB * TT * CC;  // 8388608
  const size_t NW = (size_t)CC * CC;       // 1048576

  _Float16* xh  = (_Float16*)d_ws;
  _Float16* wqh = xh + NX;
  _Float16* wkh = wqh + NW;
  _Float16* wvh = wkh + NW;
  _Float16* wph = wvh + NW;
  _Float16* qh  = wph + NW;
  _Float16* kh  = qh + NX;
  _Float16* vth = kh + NX;
  _Float16* yh  = vth + NX;
  // total ~92.3 MB of d_ws

  {
    int n4 = (int)(NX / 4);
    csa_cvt_f16<<<(n4 + 255) / 256, 256, 0, stream>>>(x, xh, n4);
    n4 = (int)(NW / 4);
    csa_cvt_f16<<<(n4 + 255) / 256, 256, 0, stream>>>(Wq, wqh, n4);
    csa_cvt_f16<<<(n4 + 255) / 256, 256, 0, stream>>>(Wk, wkh, n4);
    csa_cvt_f16<<<(n4 + 255) / 256, 256, 0, stream>>>(Wv, wvh, n4);
    csa_cvt_f16<<<(n4 + 255) / 256, 256, 0, stream>>>(Wp, wph, n4);
  }

  // Stage 1: QKV projections (3 * 64 * 8 = 1536 blocks of 128x128 tiles)
  csa_qkv<<<1536, 256, 0, stream>>>(xh, wqh, wkh, wvh, bq, bk, bv, qh, kh, vth);

  // Stage 2: attention (8192 waves, 4 waves/block share one (b,h))
  csa_attn<<<2048, 128, 0, stream>>>(qh, kh, vth, yh);

  // Stage 3: output projection (64 * 8 = 512 blocks)
  csa_proj<<<512, 256, 0, stream>>>(yh, wph, bp, out);
}
